// CLS_AttentionPool2d_26036091748982
// MI455X (gfx1250) — matrix operand order repaired and checked by reading, not compiled, not run on hardware
//
#include <hip/hip_runtime.h>
#include <hip/hip_bf16.h>
#include <math.h>

typedef __bf16 bf16_t;
typedef __attribute__((ext_vector_type(8)))  __bf16 v8bf;
typedef __attribute__((ext_vector_type(16))) __bf16 v16bf;
typedef __attribute__((ext_vector_type(8)))  float  v8f;

#define BATCH   64
#define CCH     512
#define TOK     1025
#define TROWS   1040      // 65 tiles of 16
#define NTILES  65
#define NHEADS  8
#define DH      64
#define LDST    520       // padded LDS row stride (halfs) to break bank conflicts

// combine two 16-byte chunks into a v16bf WMMA operand register set
static __device__ inline v16bf ld2(const bf16_t* p0, const bf16_t* p1) {
    v8bf a = *(const v8bf*)p0;
    v8bf b = *(const v8bf*)p1;
    v16bf r;
#pragma unroll
    for (int i = 0; i < 8; ++i) { r[i] = a[i]; r[i + 8] = b[i]; }
    return r;
}

// ---------------- kernel 1: per-(b,c) mean -> CLS row of seq ----------------
__global__ void k_mean_row0(const float* __restrict__ x, const float* __restrict__ pos,
                            bf16_t* __restrict__ seq) {
    int gtid = blockIdx.x * blockDim.x + threadIdx.x;
    int wid  = gtid >> 5;
    int lane = gtid & 31;
    if (wid >= BATCH * CCH) return;
    int b = wid >> 9, c = wid & 511;
    const float* p = x + ((size_t)(b * CCH + c) << 10);  // x[b][c][0..1023]
    float s = 0.f;
#pragma unroll 4
    for (int i = lane; i < 1024; i += 32) s += p[i];
#pragma unroll
    for (int m = 16; m >= 1; m >>= 1) s += __shfl_xor(s, m, 32);
    if (lane == 0)
        seq[(size_t)b * TROWS * CCH + c] = (bf16_t)(s * (1.0f / 1024.f) + pos[c]);
}

// ------------- kernel 2: transpose x -> seq tokens (+pos), zero pads -------------
__global__ void k_build_seq(const float* __restrict__ x, const float* __restrict__ pos,
                            bf16_t* __restrict__ seq) {
    size_t idx = (size_t)blockIdx.x * blockDim.x + threadIdx.x;
    const size_t perB = (size_t)TROWS * CCH;
    if (idx >= (size_t)BATCH * perB) return;
    int b = (int)(idx / perB);
    size_t rem = idx % perB;
    int j = (int)(rem >> 9);
    int c = (int)(rem & 511);
    if (j == 0) return;  // CLS row written by k_mean_row0
    float v = 0.f;
    if (j <= 1024)
        v = x[((size_t)(b * CCH + c) << 10) + (size_t)(j - 1)] + pos[(size_t)j * CCH + c];
    seq[idx] = (bf16_t)v;  // rows 1025..1039 zeroed (masked later anyway)
}

// ---------------- kernel 3: per-(b,h) WMMA flash attention ----------------
__global__ __launch_bounds__(128)
void k_attn(const bf16_t* __restrict__ seq,
            const float* __restrict__ Wq, const float* __restrict__ bq,
            const float* __restrict__ Wk, const float* __restrict__ bk,
            const float* __restrict__ Wv, const float* __restrict__ bv,
            float* __restrict__ out) {
    const int bh = blockIdx.x;
    const int b = bh >> 3, h = bh & 7, hbase = h * DH;
    const int tid = threadIdx.x;
    const int wid = tid >> 5, lane = tid & 31;
    const int half = lane >> 4, nn = lane & 15;

    extern __shared__ __align__(16) char smemRaw[];
    bf16_t* ldsK = (bf16_t*)smemRaw;            // 64 x LDST halfs
    bf16_t* ldsV = ldsK + DH * LDST;            // 64 x LDST halfs
    float*  fmem = (float*)(ldsV + DH * LDST);
    float* qpart = fmem;                        // 128
    float* qsL   = fmem + 128;                  // 64
    float* kbL   = fmem + 192;                  // 1
    float* wM    = fmem + 196;                  // 4
    float* wS    = fmem + 200;                  // 4
    float* wO    = fmem + 204;                  // 256

    const bf16_t* seqB = seq + (size_t)b * TROWS * CCH;

    // --- stage head's Wk/Wv slices to LDS as bf16 (row-major [dh][C], padded) ---
    for (int i = tid; i < DH * CCH; i += 128) {
        int r = i >> 9, cc = i & 511;
        ldsK[r * LDST + cc] = (bf16_t)Wk[(size_t)(hbase + r) * CCH + cc];
        ldsV[r * LDST + cc] = (bf16_t)Wv[(size_t)(hbase + r) * CCH + cc];
    }
    // --- q_h = seq[b,0] @ Wq_h^T + bq_h, pre-scaled by 1/sqrt(C) ---
    {
        int d = tid & 63, part = tid >> 6;
        const float*  wq = Wq + (size_t)(hbase + d) * CCH + part * 256;
        const bf16_t* s0 = seqB + part * 256;
        float acc = 0.f;
#pragma unroll 4
        for (int c2 = 0; c2 < 256; ++c2) acc += (float)s0[c2] * wq[c2];
        qpart[tid] = acc;
    }
    __syncthreads();
    if (tid < 64)
        qsL[tid] = (qpart[tid] + qpart[tid + 64] + bq[hbase + tid]) * 0.04419417382415922f;
    __syncthreads();
    if (tid == 0) {
        float kbv = 0.f;
        for (int n = 0; n < DH; ++n) kbv += qsL[n] * bk[hbase + n];
        kbL[0] = kbv;
    }
    __syncthreads();
    const float kb = kbL[0];
    float qv[4];
#pragma unroll
    for (int t = 0; t < 4; ++t) qv[t] = qsL[t * 16 + nn];

    float M = -INFINITY, S = 0.f;
    float O[4] = {0.f, 0.f, 0.f, 0.f};
    const v8f vzero = {0.f, 0.f, 0.f, 0.f, 0.f, 0.f, 0.f, 0.f};

    // --- stream token tiles (16 tokens each), 4 waves interleaved ---
    for (int tIdx = wid; tIdx < NTILES; tIdx += 4) {
        const int tok0 = tIdx * 16;
        v8f accK[4], accV[4];
#pragma unroll
        for (int t = 0; t < 4; ++t) { accK[t] = vzero; accV[t] = vzero; }

        const bf16_t* arow = seqB + (size_t)(tok0 + nn) * CCH + half * 8;
        // software pipeline: prefetch first A tile
        v16bf A = ld2(arow, arow + 16);
        // unroll 2: lets the A double-buffer rotate between register sets
        // (removes per-iteration v_mov copies and WAR hazard NOPs)
#pragma unroll 2
        for (int k0 = 0; k0 < CCH; k0 += 32) {
            // batch-load all 8 B tiles -> one clause of ds_load_b128
            v16bf Bk[4], Bv[4];
#pragma unroll
            for (int t = 0; t < 4; ++t) {
                const bf16_t* bp = ldsK + (t * 16 + nn) * LDST + k0 + half * 8;
                Bk[t] = ld2(bp, bp + 16);
                const bf16_t* vp = ldsV + (t * 16 + nn) * LDST + k0 + half * 8;
                Bv[t] = ld2(vp, vp + 16);
            }
            v16bf Acur = A;
            // prefetch next k-step's A under the WMMA block (wraps on last iter)
            const int kn = (k0 + 32) & (CCH - 1);
            A = ld2(arow + kn, arow + kn + 16);
            // 8 back-to-back WMMAs, no operand overwrites in between
#pragma unroll
            for (int t = 0; t < 4; ++t)
                accK[t] = __builtin_amdgcn_wmma_f32_16x16x32_bf16(
                    false, Acur, false, Bk[t], (short)0, accK[t], false, false);
#pragma unroll
            for (int t = 0; t < 4; ++t)
                accV[t] = __builtin_amdgcn_wmma_f32_16x16x32_bf16(
                    false, Acur, false, Bv[t], (short)0, accV[t], false, false);
        }

        // scores for this half's 8 tokens: m = half*8 + r, reduce over n (16 lanes)
        float sm[8];
#pragma unroll
        for (int r = 0; r < 8; ++r) {
            float s = accK[0][r] * qv[0] + accK[1][r] * qv[1]
                    + accK[2][r] * qv[2] + accK[3][r] * qv[3];
            s += __shfl_xor(s, 1, 32);
            s += __shfl_xor(s, 2, 32);
            s += __shfl_xor(s, 4, 32);
            s += __shfl_xor(s, 8, 32);
            s += kb;
            if (tok0 + half * 8 + r >= TOK) s = -INFINITY;  // mask pad tokens
            sm[r] = s;
        }
        float so[8];
#pragma unroll
        for (int r = 0; r < 8; ++r) so[r] = __shfl_xor(sm[r], 16, 32);

        float tmax = -INFINITY;
#pragma unroll
        for (int r = 0; r < 8; ++r) tmax = fmaxf(tmax, fmaxf(sm[r], so[r]));
        float newM  = fmaxf(M, tmax);
        float scale = __expf(M - newM);   // M=-inf first pass -> 0

        float p[8]; float tsum = 0.f;
#pragma unroll
        for (int r = 0; r < 8; ++r) {
            p[r] = __expf(sm[r] - newM);
            tsum += p[r] + __expf(so[r] - newM);
        }
        S = S * scale + tsum;
#pragma unroll
        for (int t = 0; t < 4; ++t) {
            float c = 0.f;
#pragma unroll
            for (int r = 0; r < 8; ++r) c += p[r] * accV[t][r];
            c += __shfl_xor(c, 16, 32);   // add other half's 8 tokens, same column
            O[t] = O[t] * scale + c;
        }
        M = newM;
    }

    // --- merge 4 waves' (M, S, O) ---
    if (half == 0) {
#pragma unroll
        for (int t = 0; t < 4; ++t) wO[wid * 64 + t * 16 + nn] = O[t];
    }
    if (lane == 0) { wM[wid] = M; wS[wid] = S; }
    __syncthreads();
    if (tid < 64) {
        float gM = fmaxf(fmaxf(wM[0], wM[1]), fmaxf(wM[2], wM[3]));
        float acc = 0.f, ss = 0.f;
#pragma unroll
        for (int w = 0; w < 4; ++w) {
            float e = __expf(wM[w] - gM);
            acc += e * wO[w * 64 + tid];
            ss  += e * wS[w];
        }
        out[(size_t)b * CCH + hbase + tid] = acc / ss + bv[hbase + tid];
    }
}

extern "C" void kernel_launch(void* const* d_in, const int* in_sizes, int n_in,
                              void* d_out, int out_size, void* d_ws, size_t ws_size,
                              hipStream_t stream) {
    (void)in_sizes; (void)n_in; (void)out_size; (void)ws_size;
    const float* x   = (const float*)d_in[0];
    const float* pos = (const float*)d_in[1];
    const float* Wq  = (const float*)d_in[2];
    const float* bq  = (const float*)d_in[3];
    const float* Wk  = (const float*)d_in[4];
    const float* bk  = (const float*)d_in[5];
    const float* Wv  = (const float*)d_in[6];
    const float* bv  = (const float*)d_in[7];
    float* out = (float*)d_out;
    bf16_t* seq = (bf16_t*)d_ws;   // [64][1040][512] bf16 = 68,157,440 bytes

    // 1) means + CLS row  (32768 waves)
    k_mean_row0<<<4096, 256, 0, stream>>>(x, pos, seq);

    // 2) token rows (transpose + pos emb + pad zeroing)
    long long total = (long long)BATCH * TROWS * CCH;
    int blocks2 = (int)((total + 255) / 256);
    k_build_seq<<<blocks2, 256, 0, stream>>>(x, pos, seq);

    // 3) fused WMMA projections + flash attention, one block per (batch, head)
    size_t smem = (size_t)2 * DH * LDST * sizeof(bf16_t) + 2048;  // ~135 KB
    hipFuncSetAttribute((const void*)k_attn,
                        hipFuncAttributeMaxDynamicSharedMemorySize, (int)smem);
    k_attn<<<BATCH * NHEADS, 128, smem, stream>>>(seq, Wq, bq, Wk, bk, Wv, bv, out);
}
